// PointNet2_35880156791482
// MI455X (gfx1250) — matrix operand order repaired and checked by reading, not compiled, not run on hardware
//
#include <hip/hip_runtime.h>
#include <hip/hip_bf16.h>

#define BB 16
#define NN 4096
#define BN_EPS 1e-5f

typedef __attribute__((ext_vector_type(16))) _Float16 v16h;
typedef __attribute__((ext_vector_type(8)))  _Float16 v8h;
typedef __attribute__((ext_vector_type(8)))  float    v8f;

static inline int cdiv(int a, int b) { return (a + b - 1) / b; }
static inline int round32(int a) { return (a + 31) & ~31; }

// ---------------------------------------------------------------------------
// xyz [B,6,N] -> pts [B,N,6] (f32 for geometry) + pts16 (f16 for features)
__global__ void k_transpose_in(const float* __restrict__ xyz,
                               float* __restrict__ pts,
                               _Float16* __restrict__ pts16) {
    int id = blockIdx.x * blockDim.x + threadIdx.x;
    int total = BB * NN * 6;
    if (id >= total) return;
    int c = id % 6;
    int n = (id / 6) % NN;
    int b = id / (6 * NN);
    float v = xyz[((size_t)b * 6 + c) * NN + n];
    pts[id] = v;
    pts16[id] = (_Float16)v;
}

// ---------------------------------------------------------------------------
// Pre-convert weights: W f32 [Cout,K] -> Wh f16 [Cout,Kpad], zero padded.
__global__ void k_prep_w(const float* __restrict__ W, int Cout, int K, int Kpad,
                         _Float16* __restrict__ Wh) {
    int id = blockIdx.x * blockDim.x + threadIdx.x;
    if (id >= Cout * Kpad) return;
    int o = id / Kpad, k = id % Kpad;
    Wh[id] = (k < K) ? (_Float16)W[(size_t)o * K + k] : (_Float16)0.0f;
}

// ---------------------------------------------------------------------------
// Farthest point sampling: one block per batch, dist in LDS.
__global__ void k_fps(const float* __restrict__ xyz, int n, int stride, int npoint,
                      int* __restrict__ fidx) {
    __shared__ float dist[4096];
    __shared__ float red_v[256];
    __shared__ int   red_i[256];
    int b = blockIdx.x;
    int t = threadIdx.x;
    const float* base = xyz + (size_t)b * n * stride;
    for (int j = t; j < n; j += blockDim.x) dist[j] = 1e10f;
    __syncthreads();
    int far = 0;
    for (int s = 0; s < npoint; ++s) {
        if (t == 0) fidx[b * npoint + s] = far;
        float cx = base[(size_t)far * stride + 0];
        float cy = base[(size_t)far * stride + 1];
        float cz = base[(size_t)far * stride + 2];
        float best = -1.0f; int bi = 0x7fffffff;
        for (int j = t; j < n; j += blockDim.x) {
            float dx = base[(size_t)j * stride + 0] - cx;
            float dy = base[(size_t)j * stride + 1] - cy;
            float dz = base[(size_t)j * stride + 2] - cz;
            float d = dx * dx + dy * dy + dz * dz;
            float nd = fminf(dist[j], d);
            dist[j] = nd;
            if (nd > best) { best = nd; bi = j; }
        }
        red_v[t] = best; red_i[t] = bi;
        __syncthreads();
        for (int off = 128; off > 0; off >>= 1) {
            if (t < off) {
                float ov = red_v[t + off]; int oi = red_i[t + off];
                if (ov > red_v[t] || (ov == red_v[t] && oi < red_i[t])) {
                    red_v[t] = ov; red_i[t] = oi;
                }
            }
            __syncthreads();
        }
        far = red_i[0];
        __syncthreads();
    }
}

// ---------------------------------------------------------------------------
__global__ void k_gather_xyz(const float* __restrict__ xyz, int n, int stride,
                             const int* __restrict__ fidx, int npoint,
                             float* __restrict__ out) {
    int id = blockIdx.x * blockDim.x + threadIdx.x;
    if (id >= BB * npoint) return;
    int b = id / npoint;
    int src = fidx[id];
    const float* p = xyz + ((size_t)b * n + src) * stride;
    out[(size_t)id * 3 + 0] = p[0];
    out[(size_t)id * 3 + 1] = p[1];
    out[(size_t)id * 3 + 2] = p[2];
}

// ---------------------------------------------------------------------------
__global__ void k_ball_query(const float* __restrict__ xyz, int n, int stride,
                             const float* __restrict__ new_xyz, int npoint,
                             float r2, int ns, int* __restrict__ gidx) {
    int id = blockIdx.x * blockDim.x + threadIdx.x;
    if (id >= BB * npoint) return;
    int b = id / npoint;
    const float* base = xyz + (size_t)b * n * stride;
    const float* c = new_xyz + (size_t)id * 3;
    int* out = gidx + (size_t)id * ns;
    int cnt = 0;
    for (int j = 0; j < n && cnt < ns; ++j) {
        float dx = base[(size_t)j * stride + 0] - c[0];
        float dy = base[(size_t)j * stride + 1] - c[1];
        float dz = base[(size_t)j * stride + 2] - c[2];
        float d = dx * dx + dy * dy + dz * dz;
        if (d <= r2) out[cnt++] = j;
    }
    int first = (cnt > 0) ? out[0] : 0;
    for (; cnt < ns; ++cnt) out[cnt] = first;
}

// ---------------------------------------------------------------------------
// Grouped SA input rows, f16, zero-padded to Kpad: [rel xyz(3) | feats(cin) | 0]
__global__ void k_group(const float* __restrict__ xyz, int n, int stridex,
                        const _Float16* __restrict__ feats, int cin,
                        const float* __restrict__ new_xyz, int npoint,
                        const int* __restrict__ gidx, int ns,
                        int c0, int ncenters, int Kpad,
                        _Float16* __restrict__ X0) {
    int row = blockIdx.x * blockDim.x + threadIdx.x;
    int rows = ncenters * ns;
    if (row >= rows) return;
    int g = c0 + row / ns;
    int j = row % ns;
    int b = g / npoint;
    int src = gidx[(size_t)g * ns + j];
    const float* p = xyz + ((size_t)b * n + src) * stridex;
    const float* c = new_xyz + (size_t)g * 3;
    _Float16* o = X0 + (size_t)row * Kpad;
    o[0] = (_Float16)(p[0] - c[0]);
    o[1] = (_Float16)(p[1] - c[1]);
    o[2] = (_Float16)(p[2] - c[2]);
    const _Float16* f = feats + ((size_t)b * n + src) * cin;
    for (int k = 0; k < cin; ++k) o[3 + k] = f[k];
    for (int k = 3 + cin; k < Kpad; ++k) o[k] = (_Float16)0.0f;
}

// ---------------------------------------------------------------------------
// WMMA GEMM + fused BN + ReLU.
// A f16 [M,Kpad], Wh f16 [Cout,Kpad] (both zero-padded, Kpad % 32 == 0).
// Inner loop: 4x global_load_b128 + 1x v_wmma_f32_16x16x32_f16, no guards.
__global__ __launch_bounds__(128)
void k_gemm(const _Float16* __restrict__ A, int M, int Kpad,
            const _Float16* __restrict__ Wh, int Cout,
            const float* __restrict__ bias,
            const float* __restrict__ gam, const float* __restrict__ bet,
            const float* __restrict__ mu, const float* __restrict__ var,
            _Float16* __restrict__ Y16, float* __restrict__ Y32, int relu) {
    int tileM = blockIdx.x;
    int tileN = blockIdx.y * 4 + threadIdx.y;
    int ntN = (Cout + 15) >> 4;
    if (tileN >= ntN) return;            // wave-uniform
    int lane = threadIdx.x;              // 0..31
    int lh = lane & 15, hi = lane >> 4;
    int m0 = tileM * 16, n0 = tileN * 16;

    const _Float16* arow = A + (size_t)(m0 + lh) * Kpad + hi * 8;
    int n = n0 + lh;
    int nw = (n < Cout) ? n : (Cout - 1);          // clamp; stores are guarded
    const _Float16* wrow = Wh + (size_t)nw * Kpad + hi * 8;

    v8f acc = {};
    for (int k0 = 0; k0 < Kpad; k0 += 32) {
        v8h a0 = *(const v8h*)(arow + k0);
        v8h a1 = *(const v8h*)(arow + k0 + 16);
        v8h b0 = *(const v8h*)(wrow + k0);
        v8h b1 = *(const v8h*)(wrow + k0 + 16);
        v16h a = __builtin_shufflevector(a0, a1, 0, 1, 2, 3, 4, 5, 6, 7,
                                         8, 9, 10, 11, 12, 13, 14, 15);
        v16h bm = __builtin_shufflevector(b0, b1, 0, 1, 2, 3, 4, 5, 6, 7,
                                          8, 9, 10, 11, 12, 13, 14, 15);
        acc = __builtin_amdgcn_wmma_f32_16x16x32_f16(
            false, a, false, bm, (short)0, acc, false, false);
    }

    if (n < Cout) {
        float bb = bias ? bias[n] : 0.0f;
        float scale = 1.0f, shift = 0.0f;
        if (gam) {
            float inv = rsqrtf(var[n] + BN_EPS);
            scale = gam[n] * inv;
            shift = bet[n] - mu[n] * scale;
        }
#pragma unroll
        for (int j = 0; j < 8; ++j) {
            int m = m0 + hi * 8 + j;
            float y = (acc[j] + bb) * scale + shift;
            if (relu) y = fmaxf(y, 0.0f);
            if (Y16) Y16[(size_t)m * Cout + n] = (_Float16)y;
            if (Y32) Y32[(size_t)m * Cout + n] = y;
        }
    }
}

// ---------------------------------------------------------------------------
__global__ void k_maxpool(const _Float16* __restrict__ Yin, int ns, int C,
                          int ncenters, _Float16* __restrict__ out) {
    int id = blockIdx.x * blockDim.x + threadIdx.x;
    if (id >= ncenters * C) return;
    int g = id / C, c = id % C;
    const _Float16* y = Yin + (size_t)g * ns * C + c;
    float m = (float)y[0];
    for (int j = 1; j < ns; ++j) m = fmaxf(m, (float)y[(size_t)j * C]);
    out[(size_t)g * C + c] = (_Float16)m;
}

// ---------------------------------------------------------------------------
__global__ void k_knn3(const float* __restrict__ xyz1, int n1, int s1,
                       const float* __restrict__ xyz2, int n2,
                       int* __restrict__ idx3, float* __restrict__ w3) {
    int id = blockIdx.x * blockDim.x + threadIdx.x;
    if (id >= BB * n1) return;
    int b = id / n1;
    const float* p = xyz1 + (size_t)id * s1;
    const float* q = xyz2 + (size_t)b * n2 * 3;
    float d0 = 3e38f, d1 = 3e38f, d2 = 3e38f;
    int i0 = 0, i1 = 0, i2 = 0;
    for (int j = 0; j < n2; ++j) {
        float dx = q[(size_t)j * 3 + 0] - p[0];
        float dy = q[(size_t)j * 3 + 1] - p[1];
        float dz = q[(size_t)j * 3 + 2] - p[2];
        float d = dx * dx + dy * dy + dz * dz;
        if (d < d0)      { d2 = d1; i2 = i1; d1 = d0; i1 = i0; d0 = d; i0 = j; }
        else if (d < d1) { d2 = d1; i2 = i1; d1 = d; i1 = j; }
        else if (d < d2) { d2 = d; i2 = j; }
    }
    float w0 = 1.0f / (d0 + 1e-8f);
    float w1 = 1.0f / (d1 + 1e-8f);
    float w2 = 1.0f / (d2 + 1e-8f);
    float s = w0 + w1 + w2;
    idx3[(size_t)id * 3 + 0] = i0;
    idx3[(size_t)id * 3 + 1] = i1;
    idx3[(size_t)id * 3 + 2] = i2;
    w3[(size_t)id * 3 + 0] = w0 / s;
    w3[(size_t)id * 3 + 1] = w1 / s;
    w3[(size_t)id * 3 + 2] = w2 / s;
}

// ---------------------------------------------------------------------------
// out[row,:] = [points1(f16) | interp of points2(f16)]; K = c1+c2 (mult of 32)
__global__ void k_interp_concat(const _Float16* __restrict__ pts1, int c1,
                                const _Float16* __restrict__ pts2, int c2, int n2,
                                const int* __restrict__ idx3,
                                const float* __restrict__ w3,
                                int n1, _Float16* __restrict__ out) {
    int K = c1 + c2;
    size_t total = (size_t)BB * n1 * K;
    size_t id = (size_t)blockIdx.x * blockDim.x + threadIdx.x;
    if (id >= total) return;
    int c = (int)(id % K);
    size_t row = id / K;
    int b = (int)(row / n1);
    float val;
    if (c < c1) {
        val = (float)pts1[row * c1 + c];
    } else {
        int cc = c - c1;
        const _Float16* f = pts2 + (size_t)b * n2 * c2;
        const int* ix = idx3 + row * 3;
        const float* w = w3 + row * 3;
        val = w[0] * (float)f[(size_t)ix[0] * c2 + cc]
            + w[1] * (float)f[(size_t)ix[1] * c2 + cc]
            + w[2] * (float)f[(size_t)ix[2] * c2 + cc];
    }
    out[row * K + c] = (_Float16)val;
}

// ---------------------------------------------------------------------------
// logits f32 [B*N,13] -> out [B,13,N]
__global__ void k_transpose_out(const float* __restrict__ logits, float* __restrict__ out) {
    int id = blockIdx.x * blockDim.x + threadIdx.x;
    int total = BB * 13 * NN;
    if (id >= total) return;
    int n = id % NN;
    int o = (id / NN) % 13;
    int b = id / (13 * NN);
    out[id] = logits[((size_t)b * NN + n) * 13 + o];
}

// ===========================================================================
struct CBN  { const float *w, *b, *g, *be, *m, *v; };
struct Conv { const float *b, *g, *be, *m, *v; const _Float16* wh; int Kpad, Cout; };

static void launch_gemm(hipStream_t st, const _Float16* A, int M, const Conv& c,
                        _Float16* Y16, float* Y32, bool relu) {
    dim3 grid(M / 16, cdiv(cdiv(c.Cout, 16), 4));
    dim3 block(32, 4);
    k_gemm<<<grid, block, 0, st>>>(A, M, c.Kpad, c.wh, c.Cout,
                                   c.b, c.g, c.be, c.m, c.v, Y16, Y32, relu ? 1 : 0);
}

static void sa_stage(hipStream_t st, const float* src_xyz, int n, int stridex,
                     const _Float16* feats, int cin, int npoint, float radius, int ns,
                     const Conv* L,
                     int* fidx, int* gidx, float* new_xyz, _Float16* pooled,
                     _Float16* bufA, _Float16* bufB, int chunk_centers) {
    k_fps<<<BB, 256, 0, st>>>(src_xyz, n, stridex, npoint, fidx);
    int tot = BB * npoint;
    k_gather_xyz<<<cdiv(tot, 256), 256, 0, st>>>(src_xyz, n, stridex, fidx, npoint, new_xyz);
    k_ball_query<<<cdiv(tot, 256), 256, 0, st>>>(src_xyz, n, stridex, new_xyz, npoint,
                                                 radius * radius, ns, gidx);
    int Kpad0 = L[0].Kpad;
    for (int c0 = 0; c0 < tot; c0 += chunk_centers) {
        int CC = (tot - c0 < chunk_centers) ? (tot - c0) : chunk_centers;
        int rows = CC * ns;
        k_group<<<cdiv(rows, 128), 128, 0, st>>>(src_xyz, n, stridex, feats, cin,
                                                 new_xyz, npoint, gidx, ns, c0, CC,
                                                 Kpad0, bufA);
        launch_gemm(st, bufA, rows, L[0], bufB, nullptr, true);
        launch_gemm(st, bufB, rows, L[1], bufA, nullptr, true);
        launch_gemm(st, bufA, rows, L[2], bufB, nullptr, true);
        k_maxpool<<<cdiv(CC * L[2].Cout, 256), 256, 0, st>>>(bufB, ns, L[2].Cout, CC,
                                                pooled + (size_t)c0 * L[2].Cout);
    }
}

static void fp_stage(hipStream_t st, const float* xyz1, int n1, int s1,
                     const float* xyz2, int n2,
                     const _Float16* pts1, int c1, const _Float16* pts2, int c2,
                     const Conv* L, int nlayers,
                     int* idx3, float* w3, _Float16* bufA, _Float16* bufB,
                     _Float16* outp) {
    int tot = BB * n1;
    k_knn3<<<cdiv(tot, 256), 256, 0, st>>>(xyz1, n1, s1, xyz2, n2, idx3, w3);
    int K0 = c1 + c2;
    size_t total = (size_t)tot * K0;
    k_interp_concat<<<(int)((total + 255) / 256), 256, 0, st>>>(pts1, c1, pts2, c2, n2,
                                                                idx3, w3, n1, bufA);
    const _Float16* cur = bufA;
    _Float16* ping[2] = { bufB, bufA };
    for (int i = 0; i < nlayers; ++i) {
        _Float16* dst = (i == nlayers - 1) ? outp : ping[i & 1];
        launch_gemm(st, cur, tot, L[i], dst, nullptr, true);
        cur = dst;
    }
}

// ===========================================================================
extern "C" void kernel_launch(void* const* d_in, const int* in_sizes, int n_in,
                              void* d_out, int out_size, void* d_ws, size_t ws_size,
                              hipStream_t stream) {
    (void)in_sizes; (void)n_in; (void)out_size; (void)ws_size;

    // ---- inputs, setup_inputs() dict insertion order ----------------------
    const float* xyz = (const float*)d_in[0];
    int cur = 1;
    auto next_cbn = [&](void) -> CBN {
        CBN p;
        p.w  = (const float*)d_in[cur++]; p.b  = (const float*)d_in[cur++];
        p.g  = (const float*)d_in[cur++]; p.be = (const float*)d_in[cur++];
        p.m  = (const float*)d_in[cur++]; p.v  = (const float*)d_in[cur++];
        return p;
    };
    CBN sa1[3], sa2[3], sa3[3], sa4[3], fp4[2], fp3[2], fp2[2], fp1[3], headp;
    for (int i = 0; i < 3; ++i) sa1[i] = next_cbn();
    for (int i = 0; i < 3; ++i) sa2[i] = next_cbn();
    for (int i = 0; i < 3; ++i) sa3[i] = next_cbn();
    for (int i = 0; i < 3; ++i) sa4[i] = next_cbn();
    for (int i = 0; i < 2; ++i) fp4[i] = next_cbn();
    for (int i = 0; i < 2; ++i) fp3[i] = next_cbn();
    for (int i = 0; i < 2; ++i) fp2[i] = next_cbn();
    for (int i = 0; i < 3; ++i) fp1[i] = next_cbn();
    headp = next_cbn();
    CBN outp_cbn;
    outp_cbn.w = (const float*)d_in[cur++];
    outp_cbn.b = (const float*)d_in[cur++];
    outp_cbn.g = nullptr; outp_cbn.be = nullptr; outp_cbn.m = nullptr; outp_cbn.v = nullptr;

    // ---- workspace bump allocator ----------------------------------------
    char* ws = (char*)d_ws;
    size_t off = 0;
    auto alloc = [&](size_t bytes) -> void* {
        void* p = ws + off;
        off += (bytes + 255) & ~(size_t)255;
        return p;
    };
    float*     pts    = (float*)alloc((size_t)BB * NN * 6 * 4);
    _Float16*  pts16  = (_Float16*)alloc((size_t)BB * NN * 6 * 2);
    float*     l1_xyz = (float*)alloc((size_t)BB * 1024 * 3 * 4);
    float*     l2_xyz = (float*)alloc((size_t)BB * 256 * 3 * 4);
    float*     l3_xyz = (float*)alloc((size_t)BB * 64 * 3 * 4);
    float*     l4_xyz = (float*)alloc((size_t)BB * 16 * 3 * 4);
    int*       fidx   = (int*)alloc((size_t)BB * 1024 * 4);
    int*       gidx   = (int*)alloc((size_t)BB * 1024 * 32 * 4);
    _Float16*  l1p    = (_Float16*)alloc((size_t)BB * 1024 * 64 * 2);
    _Float16*  l2p    = (_Float16*)alloc((size_t)BB * 256 * 128 * 2);
    _Float16*  l3p    = (_Float16*)alloc((size_t)BB * 64 * 256 * 2);
    _Float16*  l4p    = (_Float16*)alloc((size_t)BB * 16 * 512 * 2);
    _Float16*  l3f    = (_Float16*)alloc((size_t)BB * 64 * 256 * 2);
    _Float16*  l2f    = (_Float16*)alloc((size_t)BB * 256 * 256 * 2);
    _Float16*  l1f    = (_Float16*)alloc((size_t)BB * 1024 * 128 * 2);
    _Float16*  l0p    = (_Float16*)alloc((size_t)BB * NN * 128 * 2);
    int*       idx3   = (int*)alloc((size_t)BB * NN * 3 * 4);
    float*     w3     = (float*)alloc((size_t)BB * NN * 3 * 4);
    float*     logits = (float*)alloc((size_t)BB * NN * 13 * 4);
    const size_t BUFH = (size_t)8388608;               // halves (16 MB each)
    _Float16* bufA = (_Float16*)alloc(BUFH * 2);
    _Float16* bufB = (_Float16*)alloc(BUFH * 2);

    // ---- prep f16 padded weights -----------------------------------------
    auto make_conv = [&](const CBN& p, int K, int Cout) -> Conv {
        int Kpad = round32(K);
        _Float16* wh = (_Float16*)alloc((size_t)Cout * Kpad * 2);
        k_prep_w<<<cdiv(Cout * Kpad, 256), 256, 0, stream>>>(p.w, Cout, K, Kpad, wh);
        Conv c;
        c.b = p.b; c.g = p.g; c.be = p.be; c.m = p.m; c.v = p.v;
        c.wh = wh; c.Kpad = Kpad; c.Cout = Cout;
        return c;
    };
    Conv csa1[3] = { make_conv(sa1[0], 9, 32),   make_conv(sa1[1], 32, 32),
                     make_conv(sa1[2], 32, 64) };
    Conv csa2[3] = { make_conv(sa2[0], 67, 64),  make_conv(sa2[1], 64, 64),
                     make_conv(sa2[2], 64, 128) };
    Conv csa3[3] = { make_conv(sa3[0], 131, 128), make_conv(sa3[1], 128, 128),
                     make_conv(sa3[2], 128, 256) };
    Conv csa4[3] = { make_conv(sa4[0], 259, 256), make_conv(sa4[1], 256, 256),
                     make_conv(sa4[2], 256, 512) };
    Conv cfp4[2] = { make_conv(fp4[0], 768, 256), make_conv(fp4[1], 256, 256) };
    Conv cfp3[2] = { make_conv(fp3[0], 384, 256), make_conv(fp3[1], 256, 256) };
    Conv cfp2[2] = { make_conv(fp2[0], 320, 256), make_conv(fp2[1], 256, 128) };
    Conv cfp1[3] = { make_conv(fp1[0], 128, 128), make_conv(fp1[1], 128, 128),
                     make_conv(fp1[2], 128, 128) };
    Conv chead   = make_conv(headp, 128, 128);
    Conv cout    = make_conv(outp_cbn, 128, 13);

    // ---- forward ----------------------------------------------------------
    k_transpose_in<<<cdiv(BB * NN * 6, 256), 256, 0, stream>>>(xyz, pts, pts16);
    const float* l0_xyz = pts;                         // stride 6, first 3 ch

    sa_stage(stream, l0_xyz, NN,   6, pts16, 6,  1024, 0.1f, 32, csa1,
             fidx, gidx, l1_xyz, l1p, bufA, bufB, 4096);
    sa_stage(stream, l1_xyz, 1024, 3, l1p, 64,   256,  0.2f, 32, csa2,
             fidx, gidx, l2_xyz, l2p, bufA, bufB, 2048);
    sa_stage(stream, l2_xyz, 256,  3, l2p, 128,  64,   0.4f, 32, csa3,
             fidx, gidx, l3_xyz, l3p, bufA, bufB, 1024);
    sa_stage(stream, l3_xyz, 64,   3, l3p, 256,  16,   0.8f, 32, csa4,
             fidx, gidx, l4_xyz, l4p, bufA, bufB, 256);

    fp_stage(stream, l3_xyz, 64,   3, l4_xyz, 16,   l3p, 256, l4p, 512,
             cfp4, 2, idx3, w3, bufA, bufB, l3f);
    fp_stage(stream, l2_xyz, 256,  3, l3_xyz, 64,   l2p, 128, l3f, 256,
             cfp3, 2, idx3, w3, bufA, bufB, l2f);
    fp_stage(stream, l1_xyz, 1024, 3, l2_xyz, 256,  l1p, 64,  l2f, 256,
             cfp2, 2, idx3, w3, bufA, bufB, l1f);
    fp_stage(stream, l0_xyz, NN,   6, l1_xyz, 1024, nullptr, 0, l1f, 128,
             cfp1, 3, idx3, w3, bufA, bufB, l0p);

    // head conv (f16) -> classifier (f32 logits) -> (B,13,N)
    launch_gemm(stream, l0p, BB * NN, chead, bufA, nullptr, true);
    launch_gemm(stream, bufA, BB * NN, cout, nullptr, logits, false);
    k_transpose_out<<<cdiv(BB * 13 * NN, 256), 256, 0, stream>>>(logits, (float*)d_out);
}
